// GraphRes_67439576482324
// MI455X (gfx1250) — matrix operand order repaired and checked by reading, not compiled, not run on hardware
//
#include <hip/hip_runtime.h>

// ---- static problem config (matches reference) ----
#define N_NODES   102400
#define E_EDGES   819200
#define B_GR      16
#define G_CELLS   72
#define GX_CELLS  8
#define CH_FINAL  32
#define FC_IN     (G_CELLS * CH_FINAL)   // 2304
#define NUM_OUT   2

typedef __attribute__((ext_vector_type(16))) __bf16   bf16x16;
typedef __attribute__((ext_vector_type(8)))  float    f32x8;
typedef __attribute__((ext_vector_type(4)))  unsigned u32x4;
typedef __attribute__((ext_vector_type(8)))  int      i32x8;
typedef __attribute__((ext_vector_type(4)))  int      i32x4;

__device__ __forceinline__ float eluf(float v) { return v > 0.f ? v : expm1f(v); }

// monotone float<->uint encoding for atomicMax over signed floats
__device__ __forceinline__ unsigned encf(float f) {
    unsigned u = __float_as_uint(f);
    return (u & 0x80000000u) ? ~u : (u | 0x80000000u);
}
__device__ __forceinline__ float decf(unsigned u) {
    return (u & 0x80000000u) ? __uint_as_float(u ^ 0x80000000u) : __uint_as_float(~u);
}

__global__ void fill_u32(unsigned* __restrict__ p, unsigned v, int n) {
    int i = blockIdx.x * blockDim.x + threadIdx.x;
    if (i < n) p[i] = v;
}

__global__ void count_deg(const int* __restrict__ dst, int* __restrict__ cnt, int ne) {
    int e = blockIdx.x * blockDim.x + threadIdx.x;
    if (e < ne) atomicAdd(&cnt[dst[e]], 1);
}

// Pre-pack one layer's weights into the WMMA-ready LDS image: bf16, transposed
// [co][K] (K = kernel*Cin flattened), zero-padded to [CoutPad][KTOTP].
__global__ void prep_w(const float* __restrict__ w, __bf16* __restrict__ out,
                       int Cout, int KTOT, int KTOTP, int CoutPad) {
    int idx = blockIdx.x * blockDim.x + threadIdx.x;
    if (idx >= CoutPad * KTOTP) return;
    int co = idx / KTOTP, kk = idx % KTOTP;
    float v = (co < Cout && kk < KTOT) ? w[kk * Cout + co] : 0.f;
    out[idx] = (__bf16)v;
}

// Tensor Data Mover: DMA a contiguous 2D bf16 tile (the packed weight image)
// from global memory into LDS. D# per CDNA5 ISA.
// This toolchain carries the 6-arg (clang-23 / therock) builtin form.
__device__ __forceinline__ void tdm_load_weights(unsigned lds_addr, const void* gptr,
                                                 int ktotp, int coutpad) {
#if __has_builtin(__builtin_amdgcn_tensor_load_to_lds)
    unsigned long long ga = (unsigned long long)(uintptr_t)gptr;
    u32x4 g0;
    g0[0] = 1u;                                                // count=1, user D#
    g0[1] = lds_addr;                                          // lds_addr[31:0]
    g0[2] = (unsigned)(ga & 0xFFFFFFFFu);                      // global_addr[31:0]
    g0[3] = (unsigned)((ga >> 32) & 0x01FFFFFFu) | (2u << 30); // global_addr[56:32] | type=2
    i32x8 g1;
    g1[0] = 0x00010000;               // wg_mask=0, data_size=1 (2 bytes/elem)
    g1[1] = (ktotp & 0xFFFF) << 16;   // tensor_dim0[15:0]   (bits 79:48 span d1/d2)
    g1[2] = (coutpad & 0xFFFF) << 16; // tensor_dim0[31:16]=0 | tensor_dim1[15:0]
    g1[3] = (ktotp & 0xFFFF) << 16;   // tensor_dim1[31:16]=0 | tile_dim0
    g1[4] = coutpad;                  // tile_dim1 | tile_dim2=0
    g1[5] = ktotp;                    // tensor_dim0_stride[31:0] (elements)
    g1[6] = 0;                        // stride0[47:32] | tensor_dim1_stride[15:0]
    g1[7] = 0;
    i32x4 z4 = {0, 0, 0, 0};          // 2D tensor: groups 2/3 unused
    i32x8 z8 = {0, 0, 0, 0, 0, 0, 0, 0};
    __builtin_amdgcn_tensor_load_to_lds(g0, g1, z4, z4, z8, 0);
    __builtin_amdgcn_s_wait_tensorcnt(0);
#endif
}

// Fused spline-conv layer: per 16-edge tile, out = (basis ⊗ x[src]) @ Wflat via WMMA,
// scatter-added to per-dst accumulators. Cin = 1<<LC (power of two in this net).
template <int LC, int Cout>
__launch_bounds__(256)
__global__ void spline_conv(const float*  __restrict__ xin,
                            const int*    __restrict__ src,
                            const int*    __restrict__ dst,
                            const float*  __restrict__ ea,
                            const __bf16* __restrict__ wimg,  // pre-packed [CoutPad][KTOTP]
                            float*        __restrict__ acc) {
    constexpr int Cin     = 1 << LC;
    constexpr int KTOT    = 8 * Cin;            // flattened K = kernels * Cin
    constexpr int KCH     = (KTOT + 31) / 32;   // 32-wide K chunks
    constexpr int KTOTP   = KCH * 32;           // padded K
    constexpr int CoutPad = (Cout < 16) ? 16 : Cout;
    constexpr int NT      = CoutPad / 16;       // 16-wide N tiles

    __shared__ __align__(128) __bf16 Wl[CoutPad * KTOTP];

#if __has_builtin(__builtin_amdgcn_tensor_load_to_lds)
    if ((threadIdx.x >> 5) == 0) {  // one wave issues the TDM DMA + waits TENSORcnt
        tdm_load_weights((unsigned)(uintptr_t)(void*)Wl, wimg, KTOTP, CoutPad);
    }
#else
    for (int idx = threadIdx.x; idx < CoutPad * KTOTP; idx += 256)
        Wl[idx] = wimg[idx];
#endif
    __syncthreads();

    const int lane  = threadIdx.x & 31;
    const int wave  = threadIdx.x >> 5;
    const int tile  = blockIdx.x * 8 + wave;    // one wave per 16-edge tile
    const int ebase = tile * 16;
    const int m     = lane & 15;                // A row / B,C column within tile
    const int hf    = lane >> 4;                // half-wave select
    const int e     = ebase + m;

    if (blockIdx.x + 1 < gridDim.x)             // uniform: EXEC stays all-ones
        __builtin_prefetch(&src[e + 128], 0, 1);

    const int s = src[e];

    // degree-1 B-spline basis, kernel_size=2 per dim, k = k0 + 2*k1 + 4*k2
    float u0 = fminf(fmaxf(ea[e * 3 + 0], 0.f), 1.f);
    float u1 = fminf(fmaxf(ea[e * 3 + 1], 0.f), 1.f);
    float u2 = fminf(fmaxf(ea[e * 3 + 2], 0.f), 1.f);
    float bx[2] = {1.f - u0, u0}, by[2] = {1.f - u1, u1}, bz[2] = {1.f - u2, u2};
    float bas[8];
#pragma unroll
    for (int k2 = 0; k2 < 2; ++k2)
#pragma unroll
        for (int k1 = 0; k1 < 2; ++k1)
#pragma unroll
            for (int k0 = 0; k0 < 2; ++k0)
                bas[k0 + 2 * k1 + 4 * k2] = bx[k0] * by[k1] * bz[k2];

    float xj[Cin];
#pragma unroll
    for (int ci = 0; ci < Cin; ++ci) xj[ci] = xin[s * Cin + ci];

    f32x8 c[NT] = {};

#pragma unroll
    for (int kc = 0; kc < KCH; ++kc) {
        // A fragment (16x32 bf16): lanes 0-15 = rows; VGPR v holds K = {2v|8+2v} + 8*half {+0,1}.
        // Compute both half variants with compile-time indices, select by lane half.
        bf16x16 a;
#pragma unroll
        for (int v = 0; v < 8; ++v) {
#pragma unroll
            for (int p = 0; p < 2; ++p) {
                const int kb0 = ((v < 4) ? 2 * v : 8 + 2 * v) + p;
                const int kk0 = kc * 32 + kb0;
                const int kk1 = kk0 + 8;
                float e0 = (kk0 < KTOT) ? bas[kk0 >> LC] * xj[kk0 & (Cin - 1)] : 0.f;
                float e1 = (kk1 < KTOT) ? bas[kk1 >> LC] * xj[kk1 & (Cin - 1)] : 0.f;
                a[2 * v + p] = (__bf16)(hf ? e1 : e0);
            }
        }
#pragma unroll
        for (int t = 0; t < NT; ++t) {
            // B fragment (32x16 bf16): lane = column; VGPR v holds K = {2v|8+2v} + 8*half {+0,1}.
            const int co = t * 16 + m;
            bf16x16 b;
#pragma unroll
            for (int v = 0; v < 8; ++v) {
                const __bf16* wp = &Wl[co * KTOTP + kc * 32 + 8 * hf + ((v < 4) ? 2 * v : 8 + 2 * v)];
                b[2 * v]     = wp[0];
                b[2 * v + 1] = wp[1];
            }
            c[t] = __builtin_amdgcn_wmma_f32_16x16x32_bf16(
                false, a, false, b, (short)0, c[t], false, false);
        }
    }

    // C/D layout: lane = N (lane&15), VGPR r = row r + 8*half. Scatter-add by dst.
#pragma unroll
    for (int t = 0; t < NT; ++t) {
        const int co = t * 16 + m;
        if (co < Cout) {
#pragma unroll
            for (int r = 0; r < 8; ++r) {
                int d = dst[ebase + r + 8 * hf];
                atomicAdd(&acc[d * Cout + co], c[t][r]);
            }
        }
    }
}

__global__ void finalize_elu(const float* __restrict__ acc, const int* __restrict__ cnt,
                             const float* __restrict__ res, float* __restrict__ out,
                             int n, int cout) {
    int i = blockIdx.x * blockDim.x + threadIdx.x;
    if (i >= n * cout) return;
    int node = i / cout;
    float cdeg = (float)max(cnt[node], 1);
    float v = eluf(acc[i] / cdeg);
    if (res) v += res[i];
    out[i] = v;
}

__global__ void pool_max(const float* __restrict__ h, const float* __restrict__ pos,
                         const int* __restrict__ batch, unsigned* __restrict__ pooled, int n) {
    int i = blockIdx.x * blockDim.x + threadIdx.x;
    if (i >= n) return;
    int cx = (int)floorf(pos[i * 3 + 0] / 16.f);
    int cy = (int)floorf(pos[i * 3 + 1] / 12.f);
    int cl = batch[i] * G_CELLS + cy * GX_CELLS + cx;
#pragma unroll
    for (int c = 0; c < CH_FINAL; ++c)
        atomicMax(&pooled[cl * CH_FINAL + c], encf(h[i * CH_FINAL + c]));
}

__global__ void fc_kernel(const unsigned* __restrict__ pooled, const float* __restrict__ fcw,
                          float* __restrict__ out) {
    int lane = threadIdx.x & 31, wv = threadIdx.x >> 5;
    for (int o = wv; o < B_GR * NUM_OUT; o += 8) {
        int b = o / NUM_OUT, oc = o % NUM_OUT;
        float sum = 0.f;
        for (int j = lane; j < FC_IN; j += 32) {
            float p = decf(pooled[b * FC_IN + j]);
            if (!(fabsf(p) < 1e38f)) p = 0.f;   // empty cell (-inf) / non-finite -> 0
            sum += p * fcw[j * NUM_OUT + oc];
        }
#pragma unroll
        for (int off = 16; off; off >>= 1) sum += __shfl_xor(sum, off, 32);
        if (lane == 0) out[b * NUM_OUT + oc] = sum;
    }
}

extern "C" void kernel_launch(void* const* d_in, const int* in_sizes, int n_in,
                              void* d_out, int out_size, void* d_ws, size_t ws_size,
                              hipStream_t stream) {
    const float* x     = (const float*)d_in[0];
    const float* pos   = (const float*)d_in[1];
    const float* ea    = (const float*)d_in[2];
    const int*   ei    = (const int*)d_in[3];
    const int*   batch = (const int*)d_in[4];
    const float* fcw   = (const float*)d_in[5];
    const float* W[7];
    for (int i = 0; i < 7; ++i) W[i] = (const float*)d_in[6 + i];
    float* out = (float*)d_out;

    const int* srcI = ei;
    const int* dstI = ei + E_EDGES;

    // carve workspace (256B aligned)
    uintptr_t p = (uintptr_t)d_ws;
    auto carve = [&](size_t bytes) { void* r = (void*)p; p += (bytes + 255) & ~(size_t)255; return r; };
    float*    acc    = (float*)carve((size_t)N_NODES * 32 * 4);
    float*    hA     = (float*)carve((size_t)N_NODES * 32 * 4);
    float*    hB     = (float*)carve((size_t)N_NODES * 32 * 4);
    float*    hC     = (float*)carve((size_t)N_NODES * 32 * 4);
    int*      cnt    = (int*)carve((size_t)N_NODES * 4);
    unsigned* pooled = (unsigned*)carve((size_t)B_GR * G_CELLS * CH_FINAL * 4);

    // per-layer packed bf16 weight images: [CoutPad][KTOTP]
    struct LCfg { int LC, Cout; };
    const LCfg cfg[7] = {{0, 8}, {3, 16}, {4, 16}, {4, 16}, {4, 32}, {5, 32}, {5, 32}};
    __bf16* Wimg[7];
    int imgElems[7];
    for (int i = 0; i < 7; ++i) {
        int Cin = 1 << cfg[i].LC, KTOT = 8 * Cin, KTOTP = ((KTOT + 31) / 32) * 32;
        int CoutPad = cfg[i].Cout < 16 ? 16 : cfg[i].Cout;
        imgElems[i] = CoutPad * KTOTP;
        Wimg[i] = (__bf16*)carve((size_t)imgElems[i] * sizeof(__bf16));
    }

    // pack weights once per launch (deterministic)
    for (int i = 0; i < 7; ++i) {
        int Cin = 1 << cfg[i].LC, KTOT = 8 * Cin, KTOTP = ((KTOT + 31) / 32) * 32;
        int CoutPad = cfg[i].Cout < 16 ? 16 : cfg[i].Cout;
        prep_w<<<(imgElems[i] + 255) / 256, 256, 0, stream>>>(W[i], Wimg[i], cfg[i].Cout,
                                                              KTOT, KTOTP, CoutPad);
    }

    // in-degree counts (once per launch; deterministic)
    fill_u32<<<(N_NODES + 255) / 256, 256, 0, stream>>>((unsigned*)cnt, 0u, N_NODES);
    count_deg<<<E_EDGES / 256, 256, 0, stream>>>(dstI, cnt, E_EDGES);

    const int convBlocks = E_EDGES / 128;   // 8 waves/block * 16 edges/wave

#define LAYER(LCV, COUT, IN, OUT, WIMG, RES)                                                   \
    fill_u32<<<((N_NODES * (COUT)) + 255) / 256, 256, 0, stream>>>((unsigned*)acc, 0u,         \
                                                                   N_NODES * (COUT));          \
    spline_conv<LCV, COUT><<<convBlocks, 256, 0, stream>>>(IN, srcI, dstI, ea, WIMG, acc);     \
    finalize_elu<<<((N_NODES * (COUT)) + 255) / 256, 256, 0, stream>>>(acc, cnt, RES, OUT,     \
                                                                       N_NODES, COUT);

    LAYER(0, 8,  x,  hA, Wimg[0], (const float*)nullptr)   // conv1: 1 -> 8
    LAYER(3, 16, hA, hB, Wimg[1], (const float*)nullptr)   // conv2: 8 -> 16  (hB = residual)
    LAYER(4, 16, hB, hC, Wimg[2], (const float*)nullptr)   // conv3: 16 -> 16
    LAYER(4, 16, hC, hA, Wimg[3], hB)                      // conv4 + residual
    LAYER(4, 32, hA, hC, Wimg[4], (const float*)nullptr)   // conv5: 16 -> 32 (hC = residual)
    LAYER(5, 32, hC, hB, Wimg[5], (const float*)nullptr)   // conv6: 32 -> 32
    LAYER(5, 32, hB, hA, Wimg[6], hC)                      // conv7 + residual
#undef LAYER

    // voxel-grid max pool + FC
    fill_u32<<<((B_GR * G_CELLS * CH_FINAL) + 255) / 256, 256, 0, stream>>>(
        pooled, 0x007FFFFFu /* enc(-inf) */, B_GR * G_CELLS * CH_FINAL);
    pool_max<<<(N_NODES + 255) / 256, 256, 0, stream>>>(hA, pos, batch, pooled, N_NODES);
    fc_kernel<<<1, 256, 0, stream>>>(pooled, fcw, out);
}
